// StructureExtractor_8589934868
// MI455X (gfx1250) — matrix operand order repaired and checked by reading, not compiled, not run on hardware
//
#include <hip/hip_runtime.h>

typedef __attribute__((ext_vector_type(16))) __bf16 v16bf;
typedef __attribute__((ext_vector_type(8)))  __bf16 v8bf;
typedef __attribute__((ext_vector_type(8)))  float  v8f;

#define H_DIM 768
#define F_DIM 384   /* H/2 */
#define E_DIM 9
#define R_DIM 10
#define B_DIM 8
#define S_DIM 4096
#define N_ENT 64
#define P_PAIR 2016            /* N*(N-1)/2 */
#define M1 (B_DIM * S_DIM)     /* 32768 rows for entity MLP */
#define M3 (B_DIM * P_PAIR)    /* 16128 rows for relation MLP */

// ---------------------------------------------------------------------------
// Fragment helpers (CDNA5 v_wmma_f32_16x16x32_bf16, wave32).
// A (16x32 MxK) row-major in LDS: per lane two contiguous 16B chunks.
// B (32x16 KxN) staged pre-swizzled: Bsw[tile][lane][16] contiguous 32B/lane,
//   lane = (((k>>3)&1)<<4) | (n&15), elem i = ((k>>4)<<3) | (k&7)  -> 2x ds_load_b128 per frag.
// ---------------------------------------------------------------------------
__device__ inline v16bf make_frag(v8bf lo, v8bf hi) {
  return __builtin_shufflevector(lo, hi, 0, 1, 2, 3, 4, 5, 6, 7,
                                 8, 9, 10, 11, 12, 13, 14, 15);
}

__device__ inline v16bf load_a_frag(const __bf16* p, int stride) {
  const int lane = threadIdx.x & 31;
  const int kh = (lane >> 4) << 3;              // 0 or 8
  const __bf16* r = p + (lane & 15) * stride + kh;
  v8bf lo = *(const v8bf*)r;                    // K = kh..kh+7
  v8bf hi = *(const v8bf*)(r + 16);             // K = 16+kh..16+kh+7
  return make_frag(lo, hi);
}

__device__ inline v16bf load_b_frag_sw(const __bf16* tileBase) {
  const int lane = threadIdx.x & 31;
  const v8bf* p = (const v8bf*)(tileBase + lane * 16);
  return make_frag(p[0], p[1]);
}

// Stage one K-octet (8 consecutive k, one column) of a KxN chunk into Bsw.
// src row-major [K][ldsrc]; kg in 0..3 selects k = kg*8..kg*8+7 of the 32-chunk.
__device__ inline void stage_b_octet(__bf16* Bsw, const float* src, int ldsrc,
                                     int k0, int kg, int c, int ncValid) {
  v8bf v;
#pragma unroll
  for (int j = 0; j < 8; ++j)
    v[j] = (__bf16)((c < ncValid) ? src[(size_t)(k0 + kg * 8 + j) * ldsrc + c] : 0.f);
  const int nt = c >> 4;
  const int ln = ((kg & 1) << 4) | (c & 15);
  *(v8bf*)(Bsw + (size_t)(nt * 32 + ln) * 16 + (kg >> 1) * 8) = v;
}

// ---------------------------------------------------------------------------
// Kernel 1: fused  h = relu(seq@w1+b1) ;  entity_logits = h@w2+b2
// 512 blocks x 256 thr; block: 64 rows x full 384-wide h kept in LDS.
// ---------------------------------------------------------------------------
__global__ __launch_bounds__(256) void fused_entity_mlp(
    const float* __restrict__ seq, const float* __restrict__ w1,
    const float* __restrict__ b1, const float* __restrict__ w2,
    const float* __restrict__ b2, float* __restrict__ out_el) {
  extern __shared__ char smem_raw[];
  __bf16* Abuf  = (__bf16*)smem_raw;                  // 64*32 row-major (phase 1)
  __bf16* Bsw   = (__bf16*)(smem_raw + 64 * 32 * 2);  // 24 tiles swizzled (phase 1)
  __bf16* Hbuf  = (__bf16*)smem_raw;                  // 64*384 row-major (phase 2)
  __bf16* W2sw  = (__bf16*)(smem_raw + 64 * 384 * 2); // 12 chunks x 1 tile (phase 2)

  const int tid  = threadIdx.x;
  const int wave = tid >> 5;
  const int lane = tid & 31;
  const int rowBase = blockIdx.x * 64;

  const int mt     = wave & 3;             // m-tile (16 rows)
  const int ntBase = (wave >> 2) * 12;     // 12 n-tiles (192 cols) per wave

  v8f acc[12];
#pragma unroll
  for (int t = 0; t < 12; ++t) acc[t] = v8f{};

  for (int kc = 0; kc < H_DIM / 32; ++kc) {
    const int k0 = kc * 32;
    // stage A: 64x32, one 8-elem octet per thread (2x b128 load, 1x b128 store)
    {
      const int r = tid >> 2, kg = tid & 3;
      const float* g = seq + (size_t)(rowBase + r) * H_DIM + k0 + kg * 8;
      v8bf v;
#pragma unroll
      for (int j = 0; j < 8; ++j) v[j] = (__bf16)g[j];
      *(v8bf*)(Abuf + r * 32 + kg * 8) = v;
    }
    // stage B swizzled: 32x384 chunk = 1536 octets, 6 per thread
#pragma unroll
    for (int i = 0; i < 6; ++i) {
      const int o = tid + i * 256;
      stage_b_octet(Bsw, w1, F_DIM, k0, o / F_DIM, o % F_DIM, F_DIM);
    }
    if (kc + 1 < H_DIM / 32)
      __builtin_prefetch(&seq[(size_t)(rowBase + (tid >> 2)) * H_DIM + k0 + 32], 0, 1);
    __syncthreads();

    v16bf a = load_a_frag(Abuf + mt * 16 * 32, 32);
#pragma unroll
    for (int t = 0; t < 12; ++t) {
      v16bf bf = load_b_frag_sw(Bsw + (size_t)(ntBase + t) * 32 * 16);
      acc[t] = __builtin_amdgcn_wmma_f32_16x16x32_bf16(false, a, false, bf,
                                                       (short)0, acc[t], false, false);
    }
    __syncthreads();
  }

  // epilogue GEMM1: bias + relu -> Hbuf (bf16, overlays stage buffers)
  const int hi = (lane >> 4) ? 8 : 0;
  const int cit = lane & 15;
#pragma unroll
  for (int t = 0; t < 12; ++t) {
    int col = (ntBase + t) * 16 + cit;
    float bias = b1[col];
#pragma unroll
    for (int v = 0; v < 8; ++v) {
      float x = acc[t][v] + bias;
      Hbuf[(mt * 16 + hi + v) * F_DIM + col] = (__bf16)(x > 0.f ? x : 0.f);
    }
  }
  // stage w2 (padded to 16 cols) swizzled: 12 chunks x 64 octets = 768, 3/thread
#pragma unroll
  for (int i = 0; i < 3; ++i) {
    const int o = tid + i * 256;
    const int kc2 = o >> 6, w = o & 63;
    stage_b_octet(W2sw + kc2 * 512, w2, E_DIM, kc2 * 32, w >> 4, w & 15, E_DIM);
  }
  __syncthreads();

  // GEMM2: 64x384 @ 384x16 ; waves 0..3 own one m-tile each
  if (wave < 4) {
    v8f acc2 = v8f{};
#pragma unroll
    for (int kc = 0; kc < F_DIM / 32; ++kc) {
      v16bf a  = load_a_frag(Hbuf + wave * 16 * F_DIM + kc * 32, F_DIM);
      v16bf bf = load_b_frag_sw(W2sw + kc * 512);
      acc2 = __builtin_amdgcn_wmma_f32_16x16x32_bf16(false, a, false, bf,
                                                     (short)0, acc2, false, false);
    }
    if (cit < E_DIM) {
      float bias = b2[cit];
#pragma unroll
      for (int v = 0; v < 8; ++v) {
        int row = rowBase + wave * 16 + hi + v;
        out_el[(size_t)row * E_DIM + cit] = acc2[v] + bias;
      }
    }
  }
}

// ---------------------------------------------------------------------------
// Kernel 2: span mean-pooling of seq (-> entity_reprs) and entity_logits
// (-> argmax entity_types). One block per (b, n) span.
// ---------------------------------------------------------------------------
__global__ __launch_bounds__(256) void span_pool(
    const float* __restrict__ seq, const int* __restrict__ spans,
    const float* __restrict__ el, float* __restrict__ out_reprs,
    float* __restrict__ out_types) {
  const int bn  = blockIdx.x;          // b*N + n
  const int b   = bn / N_ENT;
  const int tid = threadIdx.x;
  const int start = spans[bn * 2 + 0];
  const int end   = spans[bn * 2 + 1];
  const float inv = 1.0f / fmaxf((float)(end - start), 1.0f);

  float a0 = 0.f, a1 = 0.f, a2 = 0.f, aL = 0.f;
  for (int s = start; s < end; ++s) {
    const float* row = seq + ((size_t)b * S_DIM + s) * H_DIM;
    a0 += row[tid];
    a1 += row[tid + 256];
    a2 += row[tid + 512];
    if (tid < E_DIM) aL += el[((size_t)b * S_DIM + s) * E_DIM + tid];
  }
  float* out = out_reprs + (size_t)bn * H_DIM;
  out[tid]       = a0 * inv;
  out[tid + 256] = a1 * inv;
  out[tid + 512] = a2 * inv;

  __shared__ float pl[E_DIM];
  if (tid < E_DIM) pl[tid] = aL * inv;
  __syncthreads();
  if (tid == 0) {
    int best = 0; float bv = pl[0];
    for (int e = 1; e < E_DIM; ++e)
      if (pl[e] > bv) { bv = pl[e]; best = e; }
    out_types[bn] = (float)best;
  }
}

// ---------------------------------------------------------------------------
// Kernel 3: fused  rh = relu(pair@rw1+rb1) ; relation_logits = rh@rw2+rb2
// pair rows gathered on the fly from entity_reprs via triu (i,j).
// 504 blocks x 256 thr; block: 32 pair-rows x full 768-wide rh in LDS.
// ---------------------------------------------------------------------------
__global__ __launch_bounds__(256) void fused_relation_mlp(
    const float* __restrict__ reprs, const float* __restrict__ rw1,
    const float* __restrict__ rb1, const float* __restrict__ rw2,
    const float* __restrict__ rb2, float* __restrict__ out_rl,
    float* __restrict__ out_rt) {
  extern __shared__ char smem_raw[];
  __bf16* Abuf  = (__bf16*)smem_raw;                  // 32*32 row-major (phase 1)
  __bf16* Bsw   = (__bf16*)(smem_raw + 32 * 32 * 2);  // 48 tiles swizzled (phase 1)
  __bf16* RHbuf = (__bf16*)smem_raw;                  // 32*768 row-major (phase 2)
  __bf16* W2c   = (__bf16*)(smem_raw + 32 * 768 * 2); // 1 tile swizzled (phase 2)
  int* rowsI = (int*)(smem_raw + 2 * 32 * 32 + 2 * 32 * 768);  // 32 ints
  int* rowsJ = rowsI + 32;

  const int tid  = threadIdx.x;
  const int wave = tid >> 5;
  const int lane = tid & 31;
  const int rowBase = blockIdx.x * 32;

  // decode triu (i,j) per pair-row once
  if (tid < 32) {
    int grow = rowBase + tid;
    int b = grow / P_PAIR, p = grow % P_PAIR;
    int i = 0, rem = p;
    while (rem >= (N_ENT - 1 - i)) { rem -= (N_ENT - 1 - i); ++i; }
    int j = i + 1 + rem;
    rowsI[tid] = b * N_ENT + i;
    rowsJ[tid] = b * N_ENT + j;
  }
  __syncthreads();

  const int mt     = wave & 1;            // m-tile (16 rows)
  const int ntBase = (wave >> 1) * 12;    // 12 n-tiles (192 cols) per wave

  v8f acc[12];
#pragma unroll
  for (int t = 0; t < 12; ++t) acc[t] = v8f{};

  for (int kc = 0; kc < (2 * H_DIM) / 32; ++kc) {
    const int k0 = kc * 32;
    // stage A (gather pair rows): 32x32 = 128 octets, threads 0..127
    if (tid < 128) {
      const int r = tid >> 2, kg = tid & 3;
      const int k = k0 + kg * 8;
      const float* g = (k < H_DIM)
          ? reprs + (size_t)rowsI[r] * H_DIM + k
          : reprs + (size_t)rowsJ[r] * H_DIM + (k - H_DIM);
      v8bf v;
#pragma unroll
      for (int j = 0; j < 8; ++j) v[j] = (__bf16)g[j];
      *(v8bf*)(Abuf + r * 32 + kg * 8) = v;
    }
    // stage B swizzled: 32x768 chunk = 3072 octets, 12 per thread
#pragma unroll
    for (int i = 0; i < 12; ++i) {
      const int o = tid + i * 256;
      stage_b_octet(Bsw, rw1, H_DIM, k0, o / H_DIM, o % H_DIM, H_DIM);
    }
    __syncthreads();

    v16bf a = load_a_frag(Abuf + mt * 16 * 32, 32);
#pragma unroll
    for (int t = 0; t < 12; ++t) {
      v16bf bf = load_b_frag_sw(Bsw + (size_t)(ntBase + t) * 32 * 16);
      acc[t] = __builtin_amdgcn_wmma_f32_16x16x32_bf16(false, a, false, bf,
                                                       (short)0, acc[t], false, false);
    }
    __syncthreads();
  }

  // epilogue GEMM3: bias + relu -> RHbuf bf16
  const int hi = (lane >> 4) ? 8 : 0;
  const int cit = lane & 15;
#pragma unroll
  for (int t = 0; t < 12; ++t) {
    int col = (ntBase + t) * 16 + cit;
    float bias = rb1[col];
#pragma unroll
    for (int v = 0; v < 8; ++v) {
      float x = acc[t][v] + bias;
      RHbuf[(mt * 16 + hi + v) * H_DIM + col] = (__bf16)(x > 0.f ? x : 0.f);
    }
  }
  __syncthreads();

  // GEMM4: 32x768 @ 768x16 (rw2 padded). Waves 0,1 own one m-tile each;
  // everyone cooperates on staging + barriers.
  v8f acc2 = v8f{};
  for (int kc = 0; kc < H_DIM / 32; ++kc) {
    if (tid < 64)    // 64 octets per 32x16 chunk
      stage_b_octet(W2c, rw2, R_DIM, kc * 32, tid >> 4, tid & 15, R_DIM);
    __syncthreads();
    if (wave < 2) {
      v16bf a  = load_a_frag(RHbuf + wave * 16 * H_DIM + kc * 32, H_DIM);
      v16bf bf = load_b_frag_sw(W2c);
      acc2 = __builtin_amdgcn_wmma_f32_16x16x32_bf16(false, a, false, bf,
                                                     (short)0, acc2, false, false);
    }
    __syncthreads();
  }

  if (wave < 2) {
#pragma unroll
    for (int v = 0; v < 8; ++v) {
      int prow = rowBase + wave * 16 + hi + v;
      float val = (cit < R_DIM) ? (acc2[v] + rb2[cit]) : -3.0e38f;
      if (cit < R_DIM) out_rl[(size_t)prow * R_DIM + cit] = val;
      // argmax across the 16-lane column group (ties -> lowest index)
      int bc = cit; float bv = val;
#pragma unroll
      for (int off = 8; off >= 1; off >>= 1) {
        float ov = __shfl_xor(bv, off, 32);
        int   oc = __shfl_xor(bc, off, 32);
        if (ov > bv || (ov == bv && oc < bc)) { bv = ov; bc = oc; }
      }
      if (cit == 0) out_rt[prow] = (float)bc;
    }
  }
}

// ---------------------------------------------------------------------------
extern "C" void kernel_launch(void* const* d_in, const int* in_sizes, int n_in,
                              void* d_out, int out_size, void* d_ws, size_t ws_size,
                              hipStream_t stream) {
  const float* seq   = (const float*)d_in[0];
  // d_in[1] = attention_mask (unused by the reference math)
  const int*   spans = (const int*)d_in[2];
  const float* w1    = (const float*)d_in[3];
  const float* b1    = (const float*)d_in[4];
  const float* w2    = (const float*)d_in[5];
  const float* b2    = (const float*)d_in[6];
  const float* rw1   = (const float*)d_in[7];
  const float* rb1   = (const float*)d_in[8];
  const float* rw2   = (const float*)d_in[9];
  const float* rb2   = (const float*)d_in[10];

  float* out = (float*)d_out;
  float* out_el    = out;                                       // B*S*E
  float* out_rl    = out_el + (size_t)M1 * E_DIM;               // B*P*R
  float* out_reprs = out_rl + (size_t)M3 * R_DIM;               // B*N*H
  float* out_et    = out_reprs + (size_t)B_DIM * N_ENT * H_DIM; // B*N
  float* out_rt    = out_et + B_DIM * N_ENT;                    // B*P

  const size_t smem1 = (size_t)(64 * 384 + 384 * 16) * sizeof(__bf16);       // 61440 B
  const size_t smem3 = (size_t)2 * (32 * 32 + 32 * 768) + 2 * 32 * 16 + 256; // 52480 B

  fused_entity_mlp<<<M1 / 64, 256, smem1, stream>>>(seq, w1, b1, w2, b2, out_el);
  span_pool<<<B_DIM * N_ENT, 256, 0, stream>>>(seq, spans, out_el, out_reprs, out_et);
  fused_relation_mlp<<<M3 / 32, 256, smem3, stream>>>(out_reprs, rw1, rb1, rw2, rb2,
                                                      out_rl, out_rt);
}
